// FP8GroupedLinearDeepGEMM_33243046871362
// MI455X (gfx1250) — 8-bit matrix operand order repaired and checked by reading, compiled, not run on hardware
//
#include <hip/hip_runtime.h>
#include <stdint.h>

typedef __attribute__((ext_vector_type(16))) int   v16i;
typedef __attribute__((ext_vector_type(8)))  float v8f;
typedef int v4i_vs __attribute__((vector_size(16)));

#define DEVINL __device__ __forceinline__

constexpr int   E_ = 8, M_ = 4096, K_ = 2048, N_ = 2048;
constexpr int   KB_ = K_ / 128;
constexpr int   NB_ = N_ / 128;
constexpr float FP8_MAXV = 448.0f;

DEVINL unsigned char sw_f32_to_e4m3(float f) {
  unsigned int u = __float_as_uint(f);
  unsigned char s = (unsigned char)((u >> 31) << 7);
  float a = fabsf(f);
  if (a > 448.0f) a = 448.0f;
  unsigned int bits = __float_as_uint(a);
  int e = (int)(bits >> 23) - 127;
  unsigned int m = bits & 0x7fffffu;
  unsigned int r;
  if (a == 0.0f || e < -10) {
    r = 0u;
  } else if (e < -6) {
    int shift = 20 + (-6 - e);
    unsigned int full = m | 0x800000u;
    unsigned int man  = full >> shift;
    unsigned int rem  = full & ((1u << shift) - 1u);
    unsigned int half = 1u << (shift - 1);
    if (rem > half || (rem == half && (man & 1u))) man++;
    r = man;
  } else {
    unsigned int man = m >> 20;
    unsigned int rem = m & 0xfffffu;
    if (rem > 0x80000u || (rem == 0x80000u && (man & 1u))) man++;
    unsigned int ef = (unsigned)(e + 7);
    if (man == 8u) { man = 0u; ef++; }
    r = (ef << 3) | man;
    if (r > 0x7eu) r = 0x7eu;
  }
  return (unsigned char)(r | s);
}

DEVINL unsigned int pack4_e4m3(float a, float b, float c, float d) {
#if __has_builtin(__builtin_amdgcn_cvt_pk_fp8_f32)
  int r = 0;
  r = __builtin_amdgcn_cvt_pk_fp8_f32(a, b, r, false);
  r = __builtin_amdgcn_cvt_pk_fp8_f32(c, d, r, true);
  return (unsigned int)r;
#else
  return (unsigned)sw_f32_to_e4m3(a) |
         ((unsigned)sw_f32_to_e4m3(b) << 8) |
         ((unsigned)sw_f32_to_e4m3(c) << 16) |
         ((unsigned)sw_f32_to_e4m3(d) << 24);
#endif
}

typedef __attribute__((address_space(1))) v4i_vs gv4i_t;
typedef __attribute__((address_space(3))) v4i_vs lv4i_t;

DEVINL void async_copy_16B(const void* g, void* l) {
#if __has_builtin(__builtin_amdgcn_global_load_async_to_lds_b128)
  __builtin_amdgcn_global_load_async_to_lds_b128((gv4i_t*)g, (lv4i_t*)l, 0, 0);
#else
  *(uint4*)l = *(const uint4*)g;
#endif
}

template <int N> DEVINL void wait_async() {
  asm volatile("s_wait_asynccnt %0" :: "n"(N) : "memory");
}

__global__ __launch_bounds__(256) void quant_x_kernel(
    const float* __restrict__ x, unsigned char* __restrict__ xq,
    float* __restrict__ xs) {
  int gw   = blockIdx.x * 8 + (threadIdx.x >> 5);
  int lane = threadIdx.x & 31;
  int kb   = gw & (KB_ - 1);
  int row  = gw >> 4;
  const float* p = x + (size_t)row * K_ + kb * 128 + lane * 4;
  float4 v = *(const float4*)p;
  float am = fmaxf(fmaxf(fabsf(v.x), fabsf(v.y)), fmaxf(fabsf(v.z), fabsf(v.w)));
#pragma unroll
  for (int off = 16; off > 0; off >>= 1)
    am = fmaxf(am, __shfl_xor(am, off, 32));
  float safe = fmaxf(am, 1e-4f);
  float inv  = FP8_MAXV / safe;
  unsigned int pk = pack4_e4m3(v.x * inv, v.y * inv, v.z * inv, v.w * inv);
  ((unsigned int*)xq)[((size_t)row * K_ + kb * 128) / 4 + lane] = pk;
  if (lane == 0) xs[(size_t)row * KB_ + kb] = safe / FP8_MAXV;
}

__global__ __launch_bounds__(256) void quant_w_kernel(
    const float* __restrict__ w, unsigned char* __restrict__ wq,
    float* __restrict__ wsc) {
  __shared__ float red[256];
  int b   = blockIdx.x;
  int e   = b >> 8;
  int r2  = b & 255;
  int nb  = r2 >> 4;
  int kb  = r2 & 15;
  int tid = threadIdx.x;
  const float* base = w + ((size_t)(e * N_) + nb * 128) * K_ + kb * 128;
  float am = 0.0f;
  for (int i = tid; i < 16384; i += 256) {
    int r = i >> 7, c = i & 127;
    am = fmaxf(am, fabsf(base[(size_t)r * K_ + c]));
  }
  red[tid] = am;
  __syncthreads();
  for (int s = 128; s > 0; s >>= 1) {
    if (tid < s) red[tid] = fmaxf(red[tid], red[tid + s]);
    __syncthreads();
  }
  float safe = fmaxf(red[0], 1e-6f);
  float inv  = FP8_MAXV / safe;
  if (tid == 0) wsc[(size_t)b] = safe / FP8_MAXV;
  unsigned char* qbase = wq + ((size_t)(e * N_) + nb * 128) * K_ + kb * 128;
  for (int j = tid; j < 4096; j += 256) {
    int r  = j >> 5;
    int c4 = (j & 31) << 2;
    float4 v = *(const float4*)(base + (size_t)r * K_ + c4);
    unsigned int pk = pack4_e4m3(v.x * inv, v.y * inv, v.z * inv, v.w * inv);
    *(unsigned int*)(qbase + (size_t)r * K_ + c4) = pk;
  }
}

__global__ __launch_bounds__(256) void fp8_gemm_kernel(
    const unsigned char* __restrict__ xq, const unsigned char* __restrict__ wq,
    const float* __restrict__ xs, const float* __restrict__ wsc,
    float* __restrict__ out) {
  __shared__ __align__(16) unsigned char smem[4 * 16384];

  const int bid = blockIdx.x;
  const int e   = bid >> 9;
  const int t   = bid & 511;
  const int tm  = t >> 4;
  const int tn  = t & 15;

  const int tid  = threadIdx.x;
  const int lane = tid & 31;
  const int wv   = tid >> 5;
  const int lo   = lane & 15;
  const int hi   = lane >> 4;

  const unsigned char* gA = xq + ((size_t)(e * M_) + tm * 128) * K_;
  const unsigned char* gB = wq + ((size_t)(e * N_) + tn * 128) * K_;

  auto issue = [&](int kb) {
    unsigned char* lA = smem + (kb & 1) * 16384;
    unsigned char* lB = smem + 32768 + (kb & 1) * 16384;
    const unsigned char* sA = gA + kb * 128;
    const unsigned char* sB = gB + kb * 128;
#pragma unroll
    for (int i = 0; i < 4; ++i) {
      int idx = tid + i * 256;
      int row = idx >> 3;
      int col = (idx & 7) << 4;
      async_copy_16B(sA + (size_t)row * K_ + col, lA + row * 128 + col);
    }
#pragma unroll
    for (int i = 0; i < 4; ++i) {
      int idx = tid + i * 256;
      int row = idx >> 3;
      int col = (idx & 7) << 4;
      async_copy_16B(sB + (size_t)row * K_ + col, lB + row * 128 + col);
    }
  };

  v8f acc[8];
#pragma unroll
  for (int ns = 0; ns < 8; ++ns) acc[ns] = {};

  issue(0);

  const float* xsrow = xs + ((size_t)(e * M_) + tm * 128 + wv * 16 + hi * 8) * KB_;
  const float* wsrow = wsc + ((size_t)e * NB_ + tn) * KB_;

  for (int kb = 0; kb < KB_; ++kb) {
    if (kb + 1 < KB_) { issue(kb + 1); wait_async<8>(); }
    else              { wait_async<0>(); }
    __syncthreads();

    const unsigned char* A = smem + (kb & 1) * 16384;
    const unsigned char* B = smem + 32768 + (kb & 1) * 16384;

    v16i a;
    const unsigned char* arow = A + (wv * 16 + lo) * 128 + hi * 16;
#pragma unroll
    for (int c = 0; c < 8; ++c) {
      int2 p = *(const int2*)(arow + (c >> 1) * 32 + (c & 1) * 8);
      a[2 * c] = p.x;
      a[2 * c + 1] = p.y;
    }

    const float swk = wsrow[kb];
    v8f sv;
#pragma unroll
    for (int i = 0; i < 8; ++i) sv[i] = xsrow[(size_t)i * KB_ + kb] * swk;

#pragma unroll
    for (int ns = 0; ns < 8; ++ns) {
      v16i bfr;
      const unsigned char* brow = B + (ns * 16 + lo) * 128 + hi * 16;
#pragma unroll
      for (int g = 0; g < 4; ++g) {
        int4 p = *(const int4*)(brow + g * 32);
        bfr[4 * g]     = p.x;
        bfr[4 * g + 1] = p.y;
        bfr[4 * g + 2] = p.z;
        bfr[4 * g + 3] = p.w;
      }
      v8f zero = {};
      v8f d = __builtin_amdgcn_wmma_f32_16x16x128_fp8_fp8(a, bfr, (short)0, zero,
                                                          false, false);
      acc[ns] += d * sv;
    }
    __syncthreads();
  }

  float* obase = out + ((size_t)(e * M_) + tm * 128 + wv * 16 + hi * 8) * N_ +
                 tn * 128 + lo;
#pragma unroll
  for (int ns = 0; ns < 8; ++ns)
#pragma unroll
    for (int i = 0; i < 8; ++i)
      obase[(size_t)i * N_ + ns * 16] = acc[ns][i];
}

extern "C" void kernel_launch(void* const* d_in, const int* in_sizes, int n_in,
                              void* d_out, int out_size, void* d_ws, size_t ws_size,
                              hipStream_t stream) {
  const float* x = (const float*)d_in[0];
  const float* w = (const float*)d_in[1];
  float* out = (float*)d_out;

  unsigned char* ws = (unsigned char*)d_ws;
  unsigned char* xq = ws;
  unsigned char* wq = ws + (size_t)E_ * M_ * K_;
  float* xs  = (float*)(ws + (size_t)E_ * M_ * K_ + (size_t)E_ * N_ * K_);
  float* wsc = xs + (size_t)E_ * M_ * KB_;

  quant_x_kernel<<<E_ * M_ * KB_ / 8, 256, 0, stream>>>(x, xq, xs);
  quant_w_kernel<<<E_ * NB_ * KB_, 256, 0, stream>>>(w, wq, wsc);
  fp8_gemm_kernel<<<E_ * (M_ / 128) * (N_ / 128), 256, 0, stream>>>(xq, wq, xs, wsc, out);
}
